// ZSLNet_81862076662446
// MI455X (gfx1250) — matrix operand order repaired and checked by reading, not compiled, not run on hardware
//
#include <hip/hip_runtime.h>
#include <math.h>

#define BB 8
#define NN 1024
#define DD 512
#define HH 8
#define DK 64

typedef __attribute__((ext_vector_type(16))) __bf16          v16bf;
typedef __attribute__((ext_vector_type(8)))  float           v8f;
typedef __attribute__((ext_vector_type(16))) unsigned short  v16u;
typedef __attribute__((ext_vector_type(8)))  unsigned short  v8u;

union U16cast { v16u u; v16bf b; };

__device__ __forceinline__ v16bf as_bf(v16u x) { U16cast t; t.u = x; return t.b; }

__device__ __forceinline__ v16u cat8(v8u lo, v8u hi) {
  return __builtin_shufflevector(lo, hi, 0,1,2,3,4,5,6,7,8,9,10,11,12,13,14,15);
}

__device__ __forceinline__ unsigned short f32_to_bf16(float f) {
  unsigned int u = __float_as_uint(f);
  u = u + 0x7fffu + ((u >> 16) & 1u);  // round-to-nearest-even
  return (unsigned short)(u >> 16);
}

__device__ __forceinline__ v8f v8f_zero() {
  v8f z;
#pragma unroll
  for (int i = 0; i < 8; ++i) z[i] = 0.f;
  return z;
}

__device__ __forceinline__ v8f wmma_bf16(v16bf a, v16bf b, v8f c) {
  return __builtin_amdgcn_wmma_f32_16x16x32_bf16(false, a, false, b, (short)0, c, false, false);
}

// A-operand (16x32 bf16): lane l16 holds row l16; half selects K {h*8..h*8+7, 16+h*8..}
__device__ __forceinline__ v16bf load_a_g(const unsigned short* __restrict__ base,
                                          int row, int rowstride, int k0, int hf) {
  const unsigned short* p = base + (size_t)row * rowstride + k0 + hf * 8;
  v8u lo = *(const v8u*)p;
  v8u hi = *(const v8u*)(p + 16);
  return as_bf(cat8(lo, hi));
}

// B-operand (32x16 bf16): lane l16 holds column l16; half selects K {h*8..h*8+7, 16+h*8..}
__device__ __forceinline__ v16bf load_b_g(const unsigned short* __restrict__ base,
                                          int n, int rowstride, int k0, int hf) {
  const unsigned short* p = base + (size_t)n * rowstride + k0 + hf * 8;
  v8u lo = *(const v8u*)p;
  v8u hi = *(const v8u*)(p + 16);
  return as_bf(cat8(lo, hi));
}

// ---------------- prep kernels ----------------
__global__ void k_cvt_bf16(const float* __restrict__ src, unsigned short* __restrict__ dst, int n) {
  int i = blockIdx.x * blockDim.x + threadIdx.x;
  if (i < n) dst[i] = f32_to_bf16(src[i]);
}

// W[k][n] (512x512 f32) -> Wt[n][k] bf16
__global__ void k_transpose_cvt(const float* __restrict__ W, unsigned short* __restrict__ Wt) {
  int k = blockIdx.x * blockDim.x + threadIdx.x;
  int n = blockIdx.y;
  if (k < DD) Wt[(size_t)n * DD + k] = f32_to_bf16(W[(size_t)k * DD + n]);
}

// ---------------- projection GEMM: out = X(8192x512) @ W(512x512) + bias ----------------
// 32x64 output per wave: 2 row-subtiles reuse each B operand.
// MODE 0: out bf16 [b,h,n,64]   (Q, K)
// MODE 1: out bf16 [b,h,64,n]   (V transposed)
template <int MODE>
__global__ void k_proj(const unsigned short* __restrict__ Xbf,
                       const unsigned short* __restrict__ Wt,
                       const float* __restrict__ bias,
                       unsigned short* __restrict__ out) {
  int wid = blockIdx.x;            // 2048 = 256 row-groups x 8 col-tiles
  int rowGroup = wid >> 3;
  int colTile = wid & 7;
  int lane = threadIdx.x & 31;
  int hf = lane >> 4, l16 = lane & 15;
  int rowBase = rowGroup * 32, n0 = colTile * 64;

  v8f acc[2][4];
#pragma unroll
  for (int rs = 0; rs < 2; ++rs)
#pragma unroll
    for (int t = 0; t < 4; ++t) acc[rs][t] = v8f_zero();

  for (int k0 = 0; k0 < DD; k0 += 32) {
    v16bf bmat[4];
#pragma unroll
    for (int t = 0; t < 4; ++t)
      bmat[t] = load_b_g(Wt, n0 + t * 16 + l16, DD, k0, hf);
#pragma unroll
    for (int rs = 0; rs < 2; ++rs) {
      v16bf a = load_a_g(Xbf, rowBase + rs * 16 + l16, DD, k0, hf);
#pragma unroll
      for (int t = 0; t < 4; ++t)
        acc[rs][t] = wmma_bf16(a, bmat[t], acc[rs][t]);
    }
  }

#pragma unroll
  for (int rs = 0; rs < 2; ++rs) {
#pragma unroll
    for (int t = 0; t < 4; ++t) {
      int col = n0 + t * 16 + l16;
      float bv = bias[col];
      int h = col >> 6, dk = col & 63;
#pragma unroll
      for (int r = 0; r < 8; ++r) {
        int row = rowBase + rs * 16 + r + 8 * hf;   // global token row (b*N+n)
        int bb = row >> 10, nn = row & 1023;
        unsigned short o = f32_to_bf16(acc[rs][t][r] + bv);
        if (MODE == 0)
          out[(((size_t)(bb * HH + h) * NN) + nn) * DK + dk] = o;
        else
          out[(((size_t)(bb * HH + h) * DK) + dk) * NN + nn] = o;
      }
    }
  }
}

// ---------------- fused flash attention with geometric bias ----------------
__global__ void k_attn(const unsigned short* __restrict__ Qp,   // [b,h,n,64]
                       const unsigned short* __restrict__ Kp,   // [b,h,n,64]
                       const unsigned short* __restrict__ Vt,   // [b,h,64,n]
                       const float* __restrict__ wg,            // [b,h,n,n]
                       unsigned short* __restrict__ O) {        // [b*n, 512] bf16
  __shared__ unsigned short Plds[16 * 32];

  int wid = blockIdx.x;                 // 4096 = 8*8*64
  int qt = wid & 63;
  int h = (wid >> 6) & 7;
  int b = wid >> 9;
  int lane = threadIdx.x & 31;
  int hf = lane >> 4, l16 = lane & 15;

  const unsigned short* Qb = Qp + ((((size_t)(b * HH + h)) * NN) + qt * 16) * DK;
  const unsigned short* Kb = Kp + (((size_t)(b * HH + h)) * NN) * DK;
  const unsigned short* Vb = Vt + (((size_t)(b * HH + h)) * DK) * NN;
  const float* Wb = wg + ((((size_t)(b * HH + h)) * NN) + qt * 16) * NN;

  // Q tile held in registers for whole kernel (A-operand, dk split 0..31 / 32..63)
  v16bf aQ0 = load_a_g(Qb, l16, DK, 0, hf);
  v16bf aQ1 = load_a_g(Qb, l16, DK, 32, hf);

  float m[8], ssum[8];
  v8f o[4];
#pragma unroll
  for (int r = 0; r < 8; ++r) { m[r] = -1e30f; ssum[r] = 0.f; }
#pragma unroll
  for (int t = 0; t < 4; ++t) o[t] = v8f_zero();

  for (int kc = 0; kc < NN; kc += 32) {
    // prefetch next chunk of the dominant w_g stream (gfx1250 global_prefetch_b8)
    if (kc + 32 < NN)
      __builtin_prefetch(Wb + (size_t)l16 * NN + kc + 32, 0, 0);

    // ---- S = Q K^T for 32 keys (two 16-key subtiles) ----
    v8f s0 = v8f_zero(), s1 = v8f_zero();
    {
      v16bf b00 = load_b_g(Kb, kc + l16, DK, 0, hf);
      v16bf b01 = load_b_g(Kb, kc + l16, DK, 32, hf);
      s0 = wmma_bf16(aQ0, b00, s0);
      s0 = wmma_bf16(aQ1, b01, s0);
      v16bf b10 = load_b_g(Kb, kc + 16 + l16, DK, 0, hf);
      v16bf b11 = load_b_g(Kb, kc + 16 + l16, DK, 32, hf);
      s1 = wmma_bf16(aQ0, b10, s1);
      s1 = wmma_bf16(aQ1, b11, s1);
    }

    // ---- bias + online softmax (rows live in lane-halves; 16-lane reductions) ----
    float p0[8], p1[8];
#pragma unroll
    for (int r = 0; r < 8; ++r) {
      int row = r + 8 * hf;
      float w0 = Wb[(size_t)row * NN + kc + l16];
      float w1 = Wb[(size_t)row * NN + kc + 16 + l16];
      float x0 = s0[r] * 0.125f - w0;   // 1/sqrt(64)
      float x1 = s1[r] * 0.125f - w1;
      float cm = fmaxf(x0, x1);
      cm = fmaxf(cm, __shfl_xor(cm, 1));
      cm = fmaxf(cm, __shfl_xor(cm, 2));
      cm = fmaxf(cm, __shfl_xor(cm, 4));
      cm = fmaxf(cm, __shfl_xor(cm, 8));
      float nm = fmaxf(m[r], cm);
      float alpha = __expf(m[r] - nm);
      m[r] = nm;
      float e0 = __expf(x0 - nm);
      float e1 = __expf(x1 - nm);
      float cs = e0 + e1;
      cs += __shfl_xor(cs, 1);
      cs += __shfl_xor(cs, 2);
      cs += __shfl_xor(cs, 4);
      cs += __shfl_xor(cs, 8);
      ssum[r] = ssum[r] * alpha + cs;
#pragma unroll
      for (int t = 0; t < 4; ++t) o[t][r] *= alpha;
      p0[r] = e0;
      p1[r] = e1;
    }

    // ---- transpose P (C-layout f32) -> A-layout bf16 via LDS ----
    __syncthreads();   // WAR vs previous iteration's P reads
#pragma unroll
    for (int r = 0; r < 8; ++r) {
      int row = r + 8 * hf;
      Plds[row * 32 + l16] = f32_to_bf16(p0[r]);
      Plds[row * 32 + 16 + l16] = f32_to_bf16(p1[r]);
    }
    __syncthreads();
    v16bf aP;
    {
      const unsigned short* p = &Plds[l16 * 32 + hf * 8];
      v8u lo = *(const v8u*)p;
      v8u hi = *(const v8u*)(p + 16);
      aP = as_bf(cat8(lo, hi));
    }

    // ---- O += P V  (B-operand contiguous thanks to V stored [dk][key]) ----
#pragma unroll
    for (int t = 0; t < 4; ++t) {
      v16bf bv = load_b_g(Vb, t * 16 + l16, NN, kc, hf);
      o[t] = wmma_bf16(aP, bv, o[t]);
    }
  }

  // ---- normalize and write O bf16 [row, h*64+dk] ----
  size_t orow0 = (size_t)(b * NN + qt * 16);
#pragma unroll
  for (int t = 0; t < 4; ++t) {
    int col = h * 64 + t * 16 + l16;
#pragma unroll
    for (int r = 0; r < 8; ++r) {
      int row = r + 8 * hf;
      float v = o[t][r] / ssum[r];
      O[(orow0 + row) * DD + col] = f32_to_bf16(v);
    }
  }
}

// ---------------- output projection + bias + residual (f32 out) ----------------
__global__ void k_outproj(const unsigned short* __restrict__ Obf,
                          const unsigned short* __restrict__ WoT,
                          const float* __restrict__ bo,
                          const float* __restrict__ queries,
                          float* __restrict__ X) {
  int wid = blockIdx.x;            // 2048
  int rowGroup = wid >> 3;
  int colTile = wid & 7;
  int lane = threadIdx.x & 31;
  int hf = lane >> 4, l16 = lane & 15;
  int rowBase = rowGroup * 32, n0 = colTile * 64;

  v8f acc[2][4];
#pragma unroll
  for (int rs = 0; rs < 2; ++rs)
#pragma unroll
    for (int t = 0; t < 4; ++t) acc[rs][t] = v8f_zero();

  for (int k0 = 0; k0 < DD; k0 += 32) {
    v16bf bmat[4];
#pragma unroll
    for (int t = 0; t < 4; ++t)
      bmat[t] = load_b_g(WoT, n0 + t * 16 + l16, DD, k0, hf);
#pragma unroll
    for (int rs = 0; rs < 2; ++rs) {
      v16bf a = load_a_g(Obf, rowBase + rs * 16 + l16, DD, k0, hf);
#pragma unroll
      for (int t = 0; t < 4; ++t)
        acc[rs][t] = wmma_bf16(a, bmat[t], acc[rs][t]);
    }
  }

#pragma unroll
  for (int rs = 0; rs < 2; ++rs) {
#pragma unroll
    for (int t = 0; t < 4; ++t) {
      int col = n0 + t * 16 + l16;
      float bv = bo[col];
#pragma unroll
      for (int r = 0; r < 8; ++r) {
        size_t row = (size_t)(rowBase + rs * 16 + r + 8 * hf);
        X[row * DD + col] = acc[rs][t][r] + bv + queries[row * DD + col];
      }
    }
  }
}

// ---------------- LayerNorm ----------------
__global__ void k_ln(const float* __restrict__ X, const float* __restrict__ gamma,
                     const float* __restrict__ beta, float* __restrict__ out) {
  int row = blockIdx.x;
  int tid = threadIdx.x;   // 256
  const float* x = X + (size_t)row * DD;
  float s = 0.f, s2 = 0.f;
  for (int i = tid; i < DD; i += 256) { float v = x[i]; s += v; s2 += v * v; }
  __shared__ float rs[256], rs2[256];
  rs[tid] = s; rs2[tid] = s2;
  __syncthreads();
  for (int st = 128; st > 0; st >>= 1) {
    if (tid < st) { rs[tid] += rs[tid + st]; rs2[tid] += rs2[tid + st]; }
    __syncthreads();
  }
  float mean = rs[0] * (1.f / DD);
  float var = rs2[0] * (1.f / DD) - mean * mean;
  float inv = rsqrtf(var + 1e-5f);
  for (int i = tid; i < DD; i += 256)
    out[(size_t)row * DD + i] = (x[i] - mean) * inv * gamma[i] + beta[i];
}

// ---------------- host launch ----------------
extern "C" void kernel_launch(void* const* d_in, const int* in_sizes, int n_in,
                              void* d_out, int out_size, void* d_ws, size_t ws_size,
                              hipStream_t stream) {
  const float* queries = (const float*)d_in[0];
  const float* keys    = (const float*)d_in[1];
  const float* values  = (const float*)d_in[2];
  const float* wg      = (const float*)d_in[3];
  const float* Wq = (const float*)d_in[4];
  const float* bq = (const float*)d_in[5];
  const float* Wk = (const float*)d_in[6];
  const float* bk = (const float*)d_in[7];
  const float* Wv = (const float*)d_in[8];
  const float* bv = (const float*)d_in[9];
  const float* Wo = (const float*)d_in[10];
  const float* bo = (const float*)d_in[11];
  const float* gamma = (const float*)d_in[12];
  const float* beta  = (const float*)d_in[13];
  float* outp = (float*)d_out;

  const size_t TOK = (size_t)BB * NN * DD;   // 4,194,304
  char* ws = (char*)d_ws;
  size_t off = 0;
  auto carve = [&](size_t bytes) -> void* {
    void* p = (void*)(ws + off);
    off += (bytes + 255) & ~(size_t)255;
    return p;
  };
  unsigned short* qbf = (unsigned short*)carve(TOK * 2);
  unsigned short* kbf = (unsigned short*)carve(TOK * 2);
  unsigned short* vbf = (unsigned short*)carve(TOK * 2);
  unsigned short* WqT = (unsigned short*)carve((size_t)DD * DD * 2);
  unsigned short* WkT = (unsigned short*)carve((size_t)DD * DD * 2);
  unsigned short* WvT = (unsigned short*)carve((size_t)DD * DD * 2);
  unsigned short* WoT = (unsigned short*)carve((size_t)DD * DD * 2);
  unsigned short* Qp  = (unsigned short*)carve(TOK * 2);
  unsigned short* Kp  = (unsigned short*)carve(TOK * 2);
  unsigned short* Vtp = (unsigned short*)carve(TOK * 2);
  unsigned short* Obf = (unsigned short*)carve(TOK * 2);
  float*          X   = (float*)carve(TOK * 4);
  (void)ws_size; (void)in_sizes; (void)n_in; (void)out_size;

  // 1) convert activations to bf16
  {
    int n = (int)TOK;
    dim3 g((n + 255) / 256), blk(256);
    k_cvt_bf16<<<g, blk, 0, stream>>>(queries, qbf, n);
    k_cvt_bf16<<<g, blk, 0, stream>>>(keys, kbf, n);
    k_cvt_bf16<<<g, blk, 0, stream>>>(values, vbf, n);
  }
  // 2) transpose+convert weights
  {
    dim3 g(2, DD), blk(256);
    k_transpose_cvt<<<g, blk, 0, stream>>>(Wq, WqT);
    k_transpose_cvt<<<g, blk, 0, stream>>>(Wk, WkT);
    k_transpose_cvt<<<g, blk, 0, stream>>>(Wv, WvT);
    k_transpose_cvt<<<g, blk, 0, stream>>>(Wo, WoT);
  }
  // 3) projections (WMMA, 32x64 per wave)
  {
    dim3 g(2048), blk(32);
    k_proj<0><<<g, blk, 0, stream>>>(qbf, WqT, bq, Qp);
    k_proj<0><<<g, blk, 0, stream>>>(kbf, WkT, bk, Kp);
    k_proj<1><<<g, blk, 0, stream>>>(vbf, WvT, bv, Vtp);
  }
  // 4) fused attention (WMMA flash-style, 16 q-rows per wave)
  {
    dim3 g(4096), blk(32);
    k_attn<<<g, blk, 0, stream>>>(Qp, Kp, Vtp, wg, Obf);
  }
  // 5) output projection + residual (WMMA)
  {
    dim3 g(2048), blk(32);
    k_outproj<<<g, blk, 0, stream>>>(Obf, WoT, bo, queries, X);
  }
  // 6) LayerNorm
  {
    dim3 g((unsigned)(BB * NN)), blk(256);
    k_ln<<<g, blk, 0, stream>>>(X, gamma, beta, outp);
  }
}
